// GraphSAGEEncoder_26508538151538
// MI455X (gfx1250) — matrix operand order repaired and checked by reading, not compiled, not run on hardware
//
#include <hip/hip_runtime.h>

#define N_NODES 50000
#define N_EDGES 800000
#define HID 128
#define LATENT 64
#define NLAYERS 3
#define N_GRAPHS 256
#define BN_EPS 1e-5f

typedef __attribute__((ext_vector_type(16))) _Float16 v16h;
typedef __attribute__((ext_vector_type(8)))  _Float16 v8h;
typedef __attribute__((ext_vector_type(8)))  float    v8f;

__global__ void k_zero(float* __restrict__ p, long n) {
    long i = (long)blockIdx.x * blockDim.x + threadIdx.x;
    if (i < n) p[i] = 0.0f;
}

// Pack W[k][n] (f32, row-major 128x128 per layer) into per-lane WMMA B-fragment
// order (f16): id = ((layer*8 + ct)*4 + kt)*32*16 + lane*16 + e, where
// k = kt*32 + 8*hs + (e&7) + 16*(e>>3), n = ct*16 + (lane&15), hs=lane>>4.
// Each lane's 16 halves land contiguous -> single 32B load in the GEMM.
__global__ void k_pack_w(const float* __restrict__ W, _Float16* __restrict__ P, int total) {
    int id = blockIdx.x * blockDim.x + threadIdx.x;
    if (id >= total) return;                 // total = NLAYERS*HID*HID
    int layer = id >> 14;                    // 16384 packed elems per layer
    int rem   = id & 16383;
    int e    = rem & 15;
    int lane = (rem >> 4) & 31;
    int kt   = (rem >> 9) & 3;
    int ct   = rem >> 11;                    // column tile 0..7
    int n    = ct * 16 + (lane & 15);
    int hs   = lane >> 4;
    int k    = kt * 32 + 8 * hs + (e & 7) + 16 * (e >> 3);
    P[id] = (_Float16)W[(long)layer * HID * HID + (long)k * HID + n];
}

__global__ void k_count(const int* __restrict__ dstI, float* __restrict__ cnt) {
    int e = blockIdx.x * blockDim.x + threadIdx.x;
    if (e < N_EDGES) atomicAdd(&cnt[dstI[e]], 1.0f);
}

__global__ void k_rcnt(const float* __restrict__ cnt, float* __restrict__ rcn) {
    int i = blockIdx.x * blockDim.x + threadIdx.x;
    if (i < N_NODES) rcn[i] = 1.0f / fmaxf(cnt[i], 1.0f);
}

// scatter-add h[src] into agg[dst]; one thread per (edge, 4 dims):
// b128 gather + 4 f32 atomics (aggregation set is L2-resident on MI455X).
__global__ void k_scatter(const int* __restrict__ srcI, const int* __restrict__ dstI,
                          const float* __restrict__ h, float* __restrict__ agg) {
    long idx = (long)blockIdx.x * blockDim.x + threadIdx.x;
    if (idx >= (long)N_EDGES * 32) return;
    int e = (int)(idx >> 5);
    int q = (int)(idx & 31) * 4;
    const float4 v = *(const float4*)&h[(long)srcI[e] * HID + q];
    float* a = &agg[(long)dstI[e] * HID + q];
    atomicAdd(a + 0, v.x);
    atomicAdd(a + 1, v.y);
    atomicAdd(a + 2, v.z);
    atomicAdd(a + 3, v.w);
}

// One block = 16 output rows.  8 waves; wave w computes cols [16w,16w+16).
// h_out = relu(mean @ Wl + h @ Wr + bias), K=128 via 8x wmma_f32_16x16x32_f16.
__global__ void __launch_bounds__(256)
k_sage_gemm(const float* __restrict__ agg, const float* __restrict__ rcn,
            const float* __restrict__ hin,
            const _Float16* __restrict__ wl, const _Float16* __restrict__ wr,
            const float* __restrict__ bias, float* __restrict__ hout) {
    __shared__ __align__(16) _Float16 sMean[16][HID];
    __shared__ __align__(16) _Float16 sH[16][HID];

    const int tile = blockIdx.x;          // 0..3124, rows tile*16 .. tile*16+15
    const int tid  = threadIdx.x;
    const int wave = tid >> 5;
    const int lane = tid & 31;

    const v16h* __restrict__ wlv = (const v16h*)wl;   // packed fragments
    const v16h* __restrict__ wrv = (const v16h*)wr;

    // warm L0 with this wave's weight fragments (global_prefetch_b8)
    __builtin_prefetch((const void*)&wlv[(wave * 4) * 32 + lane], 0, 0);
    __builtin_prefetch((const void*)&wrv[(wave * 4) * 32 + lane], 0, 0);

    // Stage A tiles: exactly one 8-half chunk per thread (16 rows x 16 chunks).
    // 2x global_load_b128 per stream + packed f32->f16 + 1x ds_store_b128 per stream.
    {
        const int r  = tid >> 4;          // 0..15
        const int kc = (tid & 15) * 8;    // 0,8,...,120
        long grow = (long)tile * 16 + r;
        float rc = rcn[grow];
        const float4* pa = (const float4*)&agg[grow * HID + kc];
        const float4* ph = (const float4*)&hin[grow * HID + kc];
        float4 a0 = pa[0], a1 = pa[1];
        float4 g0 = ph[0], g1 = ph[1];
        v8h mv, hv;
        mv[0] = (_Float16)(a0.x * rc); mv[1] = (_Float16)(a0.y * rc);
        mv[2] = (_Float16)(a0.z * rc); mv[3] = (_Float16)(a0.w * rc);
        mv[4] = (_Float16)(a1.x * rc); mv[5] = (_Float16)(a1.y * rc);
        mv[6] = (_Float16)(a1.z * rc); mv[7] = (_Float16)(a1.w * rc);
        hv[0] = (_Float16)g0.x; hv[1] = (_Float16)g0.y;
        hv[2] = (_Float16)g0.z; hv[3] = (_Float16)g0.w;
        hv[4] = (_Float16)g1.x; hv[5] = (_Float16)g1.y;
        hv[6] = (_Float16)g1.z; hv[7] = (_Float16)g1.w;
        *(v8h*)&sMean[r][kc] = mv;
        *(v8h*)&sH[r][kc]    = hv;
    }
    __syncthreads();

    const int m  = lane & 15;     // A-row (and B/C column N)
    const int hs = lane >> 4;     // lane-half select
    const int colb = wave * 16;

    v8f c = {};
#pragma unroll
    for (int kt = 0; kt < 4; ++kt) {
        // 16-bit A 16x32 layout per lane = two contiguous 8-half runs:
        //   [k0+8*hs, +8) and [k0+16+8*hs, +8)  ->  2x ds_load_b128 each
        const int aoff = kt * 32 + 8 * hs;
        v8h mlo = *(const v8h*)&sMean[m][aoff];
        v8h mhi = *(const v8h*)&sMean[m][aoff + 16];
        v8h hlo = *(const v8h*)&sH[m][aoff];
        v8h hhi = *(const v8h*)&sH[m][aoff + 16];
        v16h am = __builtin_shufflevector(mlo, mhi, 0, 1, 2, 3, 4, 5, 6, 7,
                                          8, 9, 10, 11, 12, 13, 14, 15);
        v16h ah = __builtin_shufflevector(hlo, hhi, 0, 1, 2, 3, 4, 5, 6, 7,
                                          8, 9, 10, 11, 12, 13, 14, 15);
        // pre-packed B fragments: one 32B contiguous load per matrix
        v16h fbl = wlv[(wave * 4 + kt) * 32 + lane];
        v16h fbr = wrv[(wave * 4 + kt) * 32 + lane];
        c = __builtin_amdgcn_wmma_f32_16x16x32_f16(false, am, false, fbl, (short)0, c, false, false);
        c = __builtin_amdgcn_wmma_f32_16x16x32_f16(false, ah, false, fbr, (short)0, c, false, false);
    }

    const float b = bias[colb + m];
#pragma unroll
    for (int r = 0; r < 8; ++r) {
        int M = r + 8 * hs;                    // C/D layout: VGPR r, lanes 16-31 -> M+8
        long row = (long)tile * 16 + M;
        float v = c[r] + b;
        hout[row * HID + colb + m] = v > 0.0f ? v : 0.0f;
    }
}

__global__ void k_pool(const float* __restrict__ h, const int* __restrict__ batch,
                       float* __restrict__ pooled) {
    long idx = (long)blockIdx.x * blockDim.x + threadIdx.x;
    if (idx >= (long)N_NODES * 32) return;
    int n = (int)(idx >> 5);
    int q = (int)(idx & 31) * 4;
    const float4 v = *(const float4*)&h[(long)n * HID + q];
    float* p = &pooled[(long)batch[n] * HID + q];
    atomicAdd(p + 0, v.x);
    atomicAdd(p + 1, v.y);
    atomicAdd(p + 2, v.z);
    atomicAdd(p + 3, v.w);
}

__global__ void k_bnstats(const float* __restrict__ pooled,
                          float* __restrict__ mu, float* __restrict__ rstd) {
    int d = threadIdx.x;                       // 128 threads
    float s = 0.0f, s2 = 0.0f;
    for (int g = 0; g < N_GRAPHS; ++g) {
        float v = pooled[g * HID + d];
        s += v; s2 += v * v;
    }
    float mean = s * (1.0f / N_GRAPHS);
    float var  = s2 * (1.0f / N_GRAPHS) - mean * mean;
    mu[d]   = mean;
    rstd[d] = rsqrtf(var + BN_EPS);
}

__global__ void k_final(const float* __restrict__ pooled, const float* __restrict__ mu,
                        const float* __restrict__ rstd, const float* __restrict__ gamma,
                        const float* __restrict__ beta, const float* __restrict__ fcW,
                        const float* __restrict__ fcb, float* __restrict__ out) {
    int g = blockIdx.x;      // 0..255
    int j = threadIdx.x;     // 0..63
    float acc = fcb[j];
    for (int d = 0; d < HID; ++d) {
        float v = (pooled[g * HID + d] - mu[d]) * rstd[d] * gamma[d] + beta[d];
        acc += v * fcW[d * LATENT + j];
    }
    out[g * LATENT + j] = acc;
}

extern "C" void kernel_launch(void* const* d_in, const int* in_sizes, int n_in,
                              void* d_out, int out_size, void* d_ws, size_t ws_size,
                              hipStream_t stream) {
    (void)in_sizes; (void)n_in; (void)out_size; (void)ws_size;

    const float* x     = (const float*)d_in[0];
    const int*   ei    = (const int*)d_in[1];
    const int*   batch = (const int*)d_in[2];
    const float* Wl    = (const float*)d_in[3];
    const float* bl    = (const float*)d_in[4];
    const float* Wr    = (const float*)d_in[5];
    const float* gamma = (const float*)d_in[6];
    const float* beta  = (const float*)d_in[7];
    const float* fcW   = (const float*)d_in[8];
    const float* fcb   = (const float*)d_in[9];
    float* out = (float*)d_out;

    const int* srcI = ei;
    const int* dstI = ei + N_EDGES;

    char* base = (char*)d_ws;
    size_t off = 0;
    auto alloc = [&](size_t bytes) -> char* {
        char* p = base + off;
        off = (off + bytes + 255) & ~(size_t)255;
        return p;
    };
    float*    agg    = (float*)alloc((size_t)N_NODES * HID * 4);
    float*    h0     = (float*)alloc((size_t)N_NODES * HID * 4);
    float*    h1     = (float*)alloc((size_t)N_NODES * HID * 4);
    float*    cnt    = (float*)alloc((size_t)N_NODES * 4);
    float*    rcn    = (float*)alloc((size_t)N_NODES * 4);
    _Float16* wl16   = (_Float16*)alloc((size_t)NLAYERS * HID * HID * 2);
    _Float16* wr16   = (_Float16*)alloc((size_t)NLAYERS * HID * HID * 2);
    float*    pooled = (float*)alloc((size_t)N_GRAPHS * HID * 4);
    float*    mu     = (float*)alloc(HID * 4);
    float*    rstd   = (float*)alloc(HID * 4);

    // weights -> f16, pre-swizzled into WMMA B-fragment order (deterministic)
    const int nw = NLAYERS * HID * HID;
    k_pack_w<<<(nw + 255) / 256, 256, 0, stream>>>(Wl, wl16, nw);
    k_pack_w<<<(nw + 255) / 256, 256, 0, stream>>>(Wr, wr16, nw);

    // in-degree counts -> reciprocal for mean aggregation
    k_zero<<<(N_NODES + 255) / 256, 256, 0, stream>>>(cnt, N_NODES);
    k_count<<<(N_EDGES + 255) / 256, 256, 0, stream>>>(dstI, cnt);
    k_rcnt<<<(N_NODES + 255) / 256, 256, 0, stream>>>(cnt, rcn);

    const float* hin = x;
    float* bufs[2] = {h0, h1};
    const long nelem = (long)N_NODES * HID;
    const long sthreads = (long)N_EDGES * 32;
    for (int layer = 0; layer < NLAYERS; ++layer) {
        float* hout = bufs[layer & 1];
        k_zero<<<(int)((nelem + 255) / 256), 256, 0, stream>>>(agg, nelem);
        k_scatter<<<(int)((sthreads + 255) / 256), 256, 0, stream>>>(srcI, dstI, hin, agg);
        k_sage_gemm<<<N_NODES / 16, 256, 0, stream>>>(
            agg, rcn, hin,
            wl16 + (long)layer * HID * HID, wr16 + (long)layer * HID * HID,
            bl + layer * HID, hout);
        hin = hout;
    }

    const long pelem = (long)N_GRAPHS * HID;
    const long pthreads = (long)N_NODES * 32;
    k_zero<<<(int)((pelem + 255) / 256), 256, 0, stream>>>(pooled, pelem);
    k_pool<<<(int)((pthreads + 255) / 256), 256, 0, stream>>>(hin, batch, pooled);
    k_bnstats<<<1, HID, 0, stream>>>(pooled, mu, rstd);
    k_final<<<N_GRAPHS, LATENT, 0, stream>>>(pooled, mu, rstd, gamma, beta, fcW, fcb, out);
}
